// biformer_layer_unet_39281770889701
// MI455X (gfx1250) — matrix operand order repaired and checked by reading, not compiled, not run on hardware
//
#include <hip/hip_runtime.h>
#include <cstdint>
#include <cstddef>

// ---------------------------------------------------------------------------
// BiFormer block (bi-level routing attention + ConvGLU) for MI455X / gfx1250.
// N=2, C=64, H=W=256, D=2, NW=8, HEADS=8, TOPK=4, KVW=4. All fp32 in/out.
// GEMMs: bf16 WMMA (v_wmma_f32_16x16x32_bf16); A tile staged by TDM
// (tensor_load_to_lds + s_wait_tensorcnt); B pre-packed per layer into WMMA
// fragment order (one 32B vector load per K-step, no branches in hot loop).
// Attention scores: f32 WMMA (v_wmma_f32_16x16x4_f32), head-dim 8 = 2 K=4 steps.
// ---------------------------------------------------------------------------

typedef __attribute__((ext_vector_type(16))) __bf16        v16bf;
typedef __attribute__((ext_vector_type(8)))  float         v8f;
typedef __attribute__((ext_vector_type(2)))  float         v2f;
typedef __attribute__((ext_vector_type(4)))  unsigned int  u32x4;
typedef __attribute__((ext_vector_type(4)))  int           i32x4;
typedef __attribute__((ext_vector_type(8)))  int           i32x8;

#ifndef __has_builtin
#define __has_builtin(x) 0
#endif

namespace bif {
constexpr int NB   = 2;
constexpr int C    = 64;
constexpr int H    = 256;
constexpr int W    = 256;
constexpr int DD   = 2;
constexpr int NW   = 8;
constexpr int P2   = NW * NW;          // 64 windows
constexpr int WS   = H / NW;           // 32 window side
constexpr int HEADS= 8;
constexpr int HD   = C / HEADS;        // 8
constexpr int TOPK = 4;
constexpr int KVW  = 4;
constexpr int KT   = TOPK * KVW * KVW; // 64 kv tokens
constexpr long long MT = (long long)NB * H * W;   // 131072 tokens
}

// ---------------------------------------------------------------------------
// TDM: load a 2D tile (rows x rowlen f32, global row stride = stride elements)
// into LDS offset 0, contiguously.  Descriptor packed per CDNA5 ISA §8.3-8.5.
// Issued by one wave; EXEC ignored by tensor ops.  Waits TENSORcnt==0.
// ---------------------------------------------------------------------------
__device__ __forceinline__ void tdm_load_tile_to_lds0(const float* gsrc,
                                                      int rowlen, int rows,
                                                      int stride_elems)
{
#if __has_builtin(__builtin_amdgcn_tensor_load_to_lds) && __has_builtin(__builtin_amdgcn_s_wait_tensorcnt)
    unsigned long long ga = (unsigned long long)(uintptr_t)gsrc;
    u32x4 g0;
    g0[0] = 1u;                                             // count=1, gather off
    g0[1] = 0u;                                             // lds_addr = base of LDS alloc
    g0[2] = (unsigned)(ga & 0xFFFFFFFFull);                 // global_addr[31:0]
    g0[3] = (unsigned)((ga >> 32) & 0x01FFFFFFull) | 0x80000000u; // addr[56:32] | type=2
    i32x8 g1;
    g1[0] = 0x20000;                                        // data_size=2 (4B), mask=0
    g1[1] = (int)((rowlen & 0xFFFF) << 16);                 // tensor_dim0[15:0]
    g1[2] = (int)(((rowlen >> 16) & 0xFFFF) | ((rows & 0xFFFF) << 16)); // dim0 hi | dim1 lo
    g1[3] = (int)(((rows >> 16) & 0xFFFF) | ((rowlen & 0xFFFF) << 16)); // dim1 hi | tile_dim0
    g1[4] = (int)(rows & 0xFFFF);                           // tile_dim1, tile_dim2=0
    g1[5] = stride_elems;                                   // tensor_dim0_stride[31:0]
    g1[6] = 0;
    g1[7] = 0;
    i32x4 z4 = {0, 0, 0, 0};
#if __clang_major__ >= 23
    i32x8 z8 = {0, 0, 0, 0, 0, 0, 0, 0};
    __builtin_amdgcn_tensor_load_to_lds(g0, g1, z4, z4, z8, 0);
#else
    __builtin_amdgcn_tensor_load_to_lds(g0, g1, z4, z4, 0);
#endif
    __builtin_amdgcn_s_wait_tensorcnt(0);
#else
    (void)gsrc; (void)rowlen; (void)rows; (void)stride_elems;
#endif
}

// ---------------------------------------------------------------------------
// Layout transposes (whole tensor fits in 192MB L2, so naive is L2-bound).
// ---------------------------------------------------------------------------
__global__ void k_nchw2nhwc(const float* __restrict__ in, float* __restrict__ out)
{
    using namespace bif;
    long long i = (long long)blockIdx.x * blockDim.x + threadIdx.x; // NHWC order
    int c = (int)(i & 63);
    long long t = i >> 6;
    int w = (int)(t & 255), h = (int)((t >> 8) & 255), n = (int)(t >> 16);
    out[i] = in[(((long long)n * C + c) * H + h) * W + w];
}

__global__ void k_nhwc2nchw(const float* __restrict__ in, float* __restrict__ out)
{
    using namespace bif;
    long long i = (long long)blockIdx.x * blockDim.x + threadIdx.x; // NCHW order
    int w = (int)(i & 255), h = (int)((i >> 8) & 255);
    int c = (int)((i >> 16) & 63), n = (int)(i >> 22);
    out[i] = in[(((long long)n * H + h) * W + w) * C + c];
}

// ---------------------------------------------------------------------------
// LayerNorm over C=64 channels, one wave per token (wave32: 2 ch / lane).
// ---------------------------------------------------------------------------
__global__ __launch_bounds__(256) void k_layernorm(const float* __restrict__ x,
                                                   const float* __restrict__ g,
                                                   const float* __restrict__ b,
                                                   float* __restrict__ out)
{
    int wave = threadIdx.x >> 5, lane = threadIdx.x & 31;
    long long tok = (long long)blockIdx.x * 8 + wave;
    const float* row = x + tok * 64;
    float a0 = row[lane], a1 = row[lane + 32];
    float s = a0 + a1;
    for (int off = 16; off; off >>= 1) s += __shfl_xor(s, off);
    float mean = s * (1.0f / 64.0f);
    float d0 = a0 - mean, d1 = a1 - mean;
    float v = d0 * d0 + d1 * d1;
    for (int off = 16; off; off >>= 1) v += __shfl_xor(v, off);
    float inv = rsqrtf(v * (1.0f / 64.0f) + 1e-6f);
    float* o = out + tok * 64;
    o[lane]      = d0 * inv * g[lane]      + b[lane];
    o[lane + 32] = d1 * inv * g[lane + 32] + b[lane + 32];
}

// ---------------------------------------------------------------------------
// Generic NHWC depthwise conv: out = [gelu](conv(src)+bias) [*mul] [+add]
// Used for pos 3x3 (+residual), LePE 5x5 (+attn), GLU dw 3x3 (gelu*vh).
// ---------------------------------------------------------------------------
__global__ void k_dwconv(const float* __restrict__ src, int ss,
                         const float* __restrict__ wgt,
                         const float* __restrict__ bias,
                         const float* __restrict__ addsrc, int as,
                         const float* __restrict__ mulsrc, int ms,
                         float* __restrict__ out, int os,
                         int cshift, int ks, int pad, int doGelu)
{
    using namespace bif;
    long long i = (long long)blockIdx.x * blockDim.x + threadIdx.x;
    int cmask = (1 << cshift) - 1;
    int c = (int)(i & cmask);
    long long t = i >> cshift;
    int w = (int)(t & 255), h = (int)((t >> 8) & 255), n = (int)(t >> 16);
    float acc = bias ? bias[c] : 0.0f;
    const float* wc = wgt + c * ks * ks;
    for (int dy = 0; dy < ks; ++dy) {
        int y = h + dy - pad;
        if ((unsigned)y >= (unsigned)H) continue;
        for (int dx = 0; dx < ks; ++dx) {
            int x = w + dx - pad;
            if ((unsigned)x >= (unsigned)W) continue;
            long long tt = ((long long)n * H + y) * W + x;
            acc += src[tt * ss + c] * wc[dy * ks + dx];
        }
    }
    if (doGelu) acc = 0.5f * acc * (1.0f + erff(acc * 0.70710678118f));
    if (mulsrc) acc *= mulsrc[t * ms + c];
    if (addsrc) acc += addsrc[t * as + c];
    out[t * os + c] = acc;
}

// ---------------------------------------------------------------------------
// Pack a weight matrix B[K,N] (f32, optionally stored transposed as [N,K])
// into bf16 WMMA B-fragment order:
//   out[((ntile*kchunks + kc32)*32 + lane)*16 + 2j + {0,1}]
//     = B[kc32*32 + (lane>>4)*16 + 2j + {0,1}][ntile*16 + (lane&15)]
// One thread per (ntile, kc32, lane) -> one aligned 32B store.
// ---------------------------------------------------------------------------
__global__ void k_packB(const float* __restrict__ B, int ldb, int btrans,
                        __bf16* __restrict__ out, int kchunks, int total)
{
    int tid = blockIdx.x * blockDim.x + threadIdx.x;
    if (tid >= total) return;
    int lane = tid & 31;
    int rest = tid >> 5;
    int kc32 = rest % kchunks;
    int nt   = rest / kchunks;
    int half = lane >> 4, mr = lane & 15;
    int ncol = nt * 16 + mr;
    v16bf b;
#pragma unroll
    for (int j = 0; j < 8; ++j) {
        int kB = kc32 * 32 + half * 16 + 2 * j;
        float b0 = btrans ? B[(long long)ncol * ldb + kB]
                          : B[(long long)kB * ldb + ncol];
        float b1 = btrans ? B[(long long)ncol * ldb + kB + 1]
                          : B[(long long)(kB + 1) * ldb + ncol];
        b[2 * j]     = (__bf16)b0;
        b[2 * j + 1] = (__bf16)b1;
    }
    ((v16bf*)out)[tid] = b;
}

// ---------------------------------------------------------------------------
// bf16 WMMA GEMM: C[M,N] = A[M,K](f32->bf16) * packedB(bf16) + bias (+res)
// Block tile 128x16, 8 waves, each wave one 16x16 f32 accumulator.
// A tile (128 x K f32) staged into LDS with one TDM transfer per block;
// B fragments are single 32-byte vector loads from the packed layout.
// ---------------------------------------------------------------------------
__global__ __launch_bounds__(256) void k_gemm_bf16(
    const float* __restrict__ A, int lda,
    const v16bf* __restrict__ packB,
    const float* __restrict__ bias,
    const float* __restrict__ res, int ldr,
    float* __restrict__ Cout, int ldc,
    int K)
{
    extern __shared__ float tileA[];                  // 128 * K floats
    const int BM = 128;
    long long bm = (long long)blockIdx.x * BM;
    int bn = blockIdx.y * 16;
    int kchunks = K >> 5;

    int wave = threadIdx.x >> 5, lane = threadIdx.x & 31;
    const v16bf* Bw = packB + ((long long)blockIdx.y * kchunks) * 32 + lane;
    __builtin_prefetch(Bw, 0, 1);                     // -> global_prefetch_b8

#if __has_builtin(__builtin_amdgcn_tensor_load_to_lds) && __has_builtin(__builtin_amdgcn_s_wait_tensorcnt)
    if (wave == 0)
        tdm_load_tile_to_lds0(A + bm * lda, K, BM, lda);
#else
    for (int i = threadIdx.x; i < BM * K; i += 256)
        tileA[i] = A[(bm + i / K) * (long long)lda + (i % K)];
#endif
    __syncthreads();

    int half = lane >> 4, mr = lane & 15;
    const float* Arow = tileA + (wave * 16 + mr) * K;
    int ncol = bn + mr;

    v8f acc = {};
    for (int kc32 = 0; kc32 < kchunks; ++kc32) {
        v16bf b = Bw[kc32 * 32];                      // one aligned 32B load
        v16bf a;
#pragma unroll
        for (int j = 0; j < 8; ++j) {
            // A (16x32 bf16) fragment: contiguous k = 16*half + s (matches packed B)
            int kA = kc32 * 32 + half * 16 + 2 * j;
            a[2 * j]     = (__bf16)Arow[kA];
            a[2 * j + 1] = (__bf16)Arow[kA + 1];
        }
#if __has_builtin(__builtin_amdgcn_wmma_f32_16x16x32_bf16)
        acc = __builtin_amdgcn_wmma_f32_16x16x32_bf16(false, a, false, b,
                                                      (short)0, acc, false, false);
#else
#pragma unroll
        for (int j = 0; j < 16; ++j) acc[0] += (float)a[j] * (float)b[j]; // placeholder
#endif
    }

    float bv = bias ? bias[ncol] : 0.0f;
    if (res) {
#pragma unroll
        for (int r = 0; r < 8; ++r) {
            long long row = bm + wave * 16 + r + 8 * half;
            Cout[row * ldc + ncol] = acc[r] + bv + res[row * ldr + ncol];
        }
    } else {
#pragma unroll
        for (int r = 0; r < 8; ++r) {
            long long row = bm + wave * 16 + r + 8 * half;
            Cout[row * ldc + ncol] = acc[r] + bv;
        }
    }
}

// ---------------------------------------------------------------------------
// Window means of q and k (32x32 window -> per-channel mean).
// ---------------------------------------------------------------------------
__global__ void k_winmean(const float* __restrict__ qkv,
                          float* __restrict__ qw, float* __restrict__ kw)
{
    using namespace bif;
    int c = threadIdx.x;                 // 64
    int p = blockIdx.x & 63, n = blockIdx.x >> 6;
    int wy = p >> 3, wx = p & 7;
    float sq = 0.0f, sk = 0.0f;
    for (int iy = 0; iy < WS; ++iy)
        for (int ix = 0; ix < WS; ++ix) {
            long long tok = ((long long)n * H + wy * WS + iy) * W + wx * WS + ix;
            const float* r = qkv + tok * 192;
            sq += r[c];
            sk += r[64 + c];
        }
    qw[(long long)blockIdx.x * 64 + c] = sq * (1.0f / 1024.0f);
    kw[(long long)blockIdx.x * 64 + c] = sk * (1.0f / 1024.0f);
}

// ---------------------------------------------------------------------------
// Region routing: logits = (qw*scale)@kw^T (64x64), hard top-4 per row.
// ---------------------------------------------------------------------------
__global__ void k_route(const float* __restrict__ qw,
                        const float* __restrict__ kw, int* __restrict__ idx)
{
    using namespace bif;
    int p = threadIdx.x, n = blockIdx.x;
    const float* q = qw + ((long long)n * P2 + p) * 64;
    float logit[P2];
    for (int j = 0; j < P2; ++j) {
        const float* k = kw + ((long long)n * P2 + j) * 64;
        float s = 0.0f;
        for (int c = 0; c < 64; ++c) s += q[c] * k[c];
        logit[j] = s * 0.125f;
    }
    unsigned long long used = 0ull;
    for (int t = 0; t < TOPK; ++t) {
        int best = 0; float bv = -3.0e38f;
        for (int j = 0; j < P2; ++j) {
            if ((used >> j) & 1ull) continue;
            if (logit[j] > bv) { bv = logit[j]; best = j; }
        }
        used |= 1ull << best;
        idx[((long long)n * P2 + p) * TOPK + t] = best;
    }
}

// ---------------------------------------------------------------------------
// Adaptive-avg-pool each 32x32 window to 4x4 for k and v -> kvp (n,p2,16,128).
// ---------------------------------------------------------------------------
__global__ void k_kvpool(const float* __restrict__ qkv, float* __restrict__ kvp)
{
    using namespace bif;
    int cc = threadIdx.x;                // 128: 0..63 k, 64..127 v
    int b = blockIdx.x;                  // n*1024 + p*16 + t
    int t = b & 15, p = (b >> 4) & 63, n = b >> 10;
    int wy = p >> 3, wx = p & 7, ky = t >> 2, kx = t & 3;
    int col = cc < 64 ? 64 + cc : 128 + (cc - 64);
    float s = 0.0f;
    for (int iy = 0; iy < 8; ++iy)
        for (int ix = 0; ix < 8; ++ix) {
            long long tok = ((long long)n * H + wy * WS + ky * 8 + iy) * W
                          + wx * WS + kx * 8 + ix;
            s += qkv[tok * 192 + col];
        }
    kvp[(long long)b * 128 + cc] = s * (1.0f / 64.0f);
}

// ---------------------------------------------------------------------------
// Routed attention. Block = (n, window, 128-query chunk); 8 waves x 16 queries.
// Gathered top-k K/V (64 tokens x 64 ch) in LDS; per head:
//   S(16x64) via v_wmma_f32_16x16x4_f32 (2 chained K=4 steps = hd 8),
//   shuffle softmax over the 64 kv, P->LDS, AV via per-lane dot with Vs.
// ---------------------------------------------------------------------------
__global__ __launch_bounds__(256) void k_attn(const float* __restrict__ qkv,
                                              const float* __restrict__ kvp,
                                              const int* __restrict__ ridx,
                                              float* __restrict__ att)
{
    using namespace bif;
    extern __shared__ float smem[];
    float* Ks = smem;                    // 64 x 64
    float* Vs = smem + KT * 64;          // 64 x 64
    float* Ps = smem + 2 * KT * 64;      // 8 waves x 16 x 64

    int qc = blockIdx.x & 7;
    int p  = (blockIdx.x >> 3) & 63;
    int n  = blockIdx.x >> 9;
    int wy = p >> 3, wx = p & 7;
    const int* ri = ridx + ((long long)n * P2 + p) * TOPK;

    for (int i = threadIdx.x; i < KT * 64; i += 256) {
        int kv = i >> 6, c = i & 63;
        int slot = kv >> 4, tk = kv & 15;
        long long base = (((long long)n * P2 + ri[slot]) * 16 + tk) * 128;
        Ks[i] = kvp[base + c];
        Vs[i] = kvp[base + 64 + c];
    }
    __syncthreads();

    int wave = threadIdx.x >> 5, lane = threadIdx.x & 31;
    int half = lane >> 4, mr = lane & 15;
    int q0 = qc * 128 + wave * 16;
    int qi = q0 + mr;
    long long tokA = ((long long)n * H + wy * WS + (qi >> 5)) * W + wx * WS + (qi & 31);
    const float* Qrow = qkv + tokA * 192;          // q = cols 0..63
    float* Pw = Ps + wave * (16 * 64);
    const float scale = 0.125f;

    for (int h = 0; h < HEADS; ++h) {
        v8f s[4] = {};
#if __has_builtin(__builtin_amdgcn_wmma_f32_16x16x4_f32)
#pragma unroll
        for (int kk = 0; kk < HD; kk += 4) {
            v2f a;
            a[0] = Qrow[h * HD + kk + half * 2 + 0] * scale;
            a[1] = Qrow[h * HD + kk + half * 2 + 1] * scale;
#pragma unroll
            for (int nb = 0; nb < 4; ++nb) {
                v2f b;
                b[0] = Ks[(nb * 16 + mr) * 64 + h * HD + kk + half * 2 + 0];
                b[1] = Ks[(nb * 16 + mr) * 64 + h * HD + kk + half * 2 + 1];
                s[nb] = __builtin_amdgcn_wmma_f32_16x16x4_f32(false, a, false, b,
                                                              (short)0, s[nb],
                                                              false, false);
            }
        }
#else
        for (int r = 0; r < 8; ++r) {
            int qrow = q0 + r + 8 * half;
            long long tr = ((long long)n * H + wy * WS + (qrow >> 5)) * W
                         + wx * WS + (qrow & 31);
            const float* Qr = qkv + tr * 192;
            for (int nb = 0; nb < 4; ++nb) {
                float acc = 0.0f;
                for (int c = 0; c < HD; ++c)
                    acc += Qr[h * HD + c] * Ks[(nb * 16 + mr) * 64 + h * HD + c];
                s[nb][r] = acc * scale;
            }
        }
#endif
        // row softmax over 64 kv (cols live in lanes 0-15 of each half + 4 frags)
#pragma unroll
        for (int r = 0; r < 8; ++r) {
            float m = fmaxf(fmaxf(s[0][r], s[1][r]), fmaxf(s[2][r], s[3][r]));
            for (int off = 8; off; off >>= 1) m = fmaxf(m, __shfl_xor(m, off));
            float e0 = __expf(s[0][r] - m), e1 = __expf(s[1][r] - m);
            float e2 = __expf(s[2][r] - m), e3 = __expf(s[3][r] - m);
            float sum = e0 + e1 + e2 + e3;
            for (int off = 8; off; off >>= 1) sum += __shfl_xor(sum, off);
            float inv = 1.0f / sum;
            int row = r + 8 * half;
            Pw[row * 64 +  0 + mr] = e0 * inv;
            Pw[row * 64 + 16 + mr] = e1 * inv;
            Pw[row * 64 + 32 + mr] = e2 * inv;
            Pw[row * 64 + 48 + mr] = e3 * inv;
        }
        // O = P @ V_h : each lane owns 4 of the 16x8 outputs (same-wave LDS
        // store->load is in-order per DScnt rules).
#pragma unroll
        for (int e = 0; e < 4; ++e) {
            int oi = lane * 4 + e;
            int qq = oi >> 3, d = oi & 7;
            float acc = 0.0f;
            for (int kv = 0; kv < KT; ++kv)
                acc += Pw[qq * 64 + kv] * Vs[kv * 64 + h * HD + d];
            int qidx = q0 + qq;
            long long tok = ((long long)n * H + wy * WS + (qidx >> 5)) * W
                          + wx * WS + (qidx & 31);
            att[tok * 64 + h * HD + d] = acc;
        }
    }
}

// ---------------------------------------------------------------------------
// Host-side orchestration.
// ---------------------------------------------------------------------------
extern "C" void kernel_launch(void* const* d_in, const int* in_sizes, int n_in,
                              void* d_out, int out_size, void* d_ws, size_t ws_size,
                              hipStream_t stream)
{
    using namespace bif;
    (void)in_sizes; (void)n_in; (void)out_size; (void)ws_size;

    const float* x_in = (const float*)d_in[0];
    float* ws = (float*)d_ws;

    // workspace layout (floats)
    const long long SZ64  = MT * 64;
    const long long SZ128 = MT * 128;
    const long long SZ192 = MT * 192;
    const long long SZ256 = MT * 256;
    float*  B0  = ws;                    // token x 64 (ping)
    float*  B1  = B0  + SZ64;            // token x 64 (pong)
    float*  QKV = B1  + SZ64;            // token x 192
    float*  ATT = QKV + SZ192;           // token x 64
    float*  T   = ATT + SZ64;            // token x 256 (fc1: xh|vh)
    float*  U   = T   + SZ256;           // token x 128 (pw out)
    float*  V2  = U   + SZ128;           // token x 128 (dw*gelu*vh)
    float*  QW  = V2  + SZ128;           // 2*64*64
    float*  KW  = QW  + NB * P2 * 64;
    float*  KVP = KW  + NB * P2 * 64;    // 2*64*16*128
    int*    IDX = (int*)(KVP + (long long)NB * P2 * 16 * 128);
    __bf16* PB  = (__bf16*)(IDX + NB * P2 * TOPK);  // packed weights, <=32KB

    const int nblk64  = (int)(SZ64  / 256);   // elementwise grids
    const int nblk128 = (int)(SZ128 / 256);
    const int lnblk   = (int)(MT / 8);
    const int gmx     = (int)(MT / 128);      // GEMM M tiles

    // helper: pack-then-GEMM launch pair
    auto gemm = [&](const float* A, int lda, const float* Bm, int ldb, int btrans,
                    const float* bias, const float* res, int ldr,
                    float* Cm, int ldc, int K, int N) {
        int kch = K >> 5;
        int total = (N / 16) * kch * 32;
        k_packB<<<(total + 255) / 256, 256, 0, stream>>>(Bm, ldb, btrans, PB,
                                                         kch, total);
        k_gemm_bf16<<<dim3(gmx, N / 16), 256, (size_t)128 * K * 4, stream>>>(
            A, lda, (const v16bf*)PB, bias, res, ldr, Cm, ldc, K);
    };

    k_nchw2nhwc<<<nblk64, 256, 0, stream>>>(x_in, B0);

    float* cur = B0;
    float* alt = B1;
    for (int i = 0; i < DD; ++i) {
        const float* pos_w  = (const float*)d_in[1]  + (long long)i * C * 9;
        const float* pos_b  = (const float*)d_in[2]  + (long long)i * C;
        const float* ln1_g  = (const float*)d_in[3]  + (long long)i * C;
        const float* ln1_b  = (const float*)d_in[4]  + (long long)i * C;
        const float* qkv_w  = (const float*)d_in[5]  + (long long)i * C * 192;
        const float* qkv_b  = (const float*)d_in[6]  + (long long)i * 192;
        const float* lepe_w = (const float*)d_in[7]  + (long long)i * C * 25;
        const float* lepe_b = (const float*)d_in[8]  + (long long)i * C;
        const float* wo_w   = (const float*)d_in[9]  + (long long)i * C * C;
        const float* wo_b   = (const float*)d_in[10] + (long long)i * C;
        const float* ln2_g  = (const float*)d_in[11] + (long long)i * C;
        const float* ln2_b  = (const float*)d_in[12] + (long long)i * C;
        const float* fc1_w  = (const float*)d_in[13] + (long long)i * C * 256;
        const float* fc1_b  = (const float*)d_in[14] + (long long)i * 256;
        const float* pw_w   = (const float*)d_in[15] + (long long)i * 128 * 128;
        const float* dw_w   = (const float*)d_in[16] + (long long)i * 128 * 9;
        const float* dw_b   = (const float*)d_in[17] + (long long)i * 128;
        const float* fc2_w  = (const float*)d_in[18] + (long long)i * 128 * 64;
        const float* fc2_b  = (const float*)d_in[19] + (long long)i * 64;

        // x = x + dwconv3x3(x) + pos_b    (alt <- cur)
        k_dwconv<<<nblk64, 256, 0, stream>>>(cur, 64, pos_w, pos_b,
                                             cur, 64, nullptr, 0,
                                             alt, 64, 6, 3, 1, 0);
        // ln1 (cur <- LN(alt))
        k_layernorm<<<lnblk, 256, 0, stream>>>(alt, ln1_g, ln1_b, cur);
        // qkv = ln1 @ qkv_w + qkv_b
        gemm(cur, 64, qkv_w, 192, 0, qkv_b, nullptr, 0, QKV, 192, 64, 192);
        // routing
        k_winmean<<<NB * P2, 64, 0, stream>>>(QKV, QW, KW);
        k_route<<<NB, P2, 0, stream>>>(QW, KW, IDX);
        k_kvpool<<<NB * P2 * 16, 128, 0, stream>>>(QKV, KVP);
        // attention
        k_attn<<<NB * P2 * 8, 256, (2 * KT * 64 + 8 * 16 * 64) * 4, stream>>>(
            QKV, KVP, IDX, ATT);
        // ATT += dwconv5x5(v) + lepe_b
        k_dwconv<<<nblk64, 256, 0, stream>>>(QKV + 128, 192, lepe_w, lepe_b,
                                             ATT, 64, nullptr, 0,
                                             ATT, 64, 6, 5, 2, 0);
        // y1 = ATT @ wo_w + wo_b + x   (cur <- result, residual = alt)
        gemm(ATT, 64, wo_w, 64, 0, wo_b, alt, 64, cur, 64, 64, 64);
        // ln2 (alt <- LN(cur))
        k_layernorm<<<lnblk, 256, 0, stream>>>(cur, ln2_g, ln2_b, alt);
        // fc1: T = ln2 @ fc1_w + fc1_b  (xh = T[:, :128], vh = T[:, 128:])
        gemm(alt, 64, fc1_w, 256, 0, fc1_b, nullptr, 0, T, 256, 64, 256);
        // pw: U = xh @ pw_w^T (conv1x1, no bias); pw_w is OI -> transposed read
        gemm(T, 256, pw_w, 128, 1, nullptr, nullptr, 0, U, 128, 128, 128);
        // V2 = gelu(dwconv3x3(U) + dw_b) * vh
        k_dwconv<<<nblk128, 256, 0, stream>>>(U, 128, dw_w, dw_b,
                                              nullptr, 0, T + 128, 256,
                                              V2, 128, 7, 3, 1, 1);
        // x_new = V2 @ fc2_w + fc2_b + y1   (alt <- result, residual = cur)
        gemm(V2, 128, fc2_w, 64, 0, fc2_b, cur, 64, alt, 64, 128, 64);

        float* tmp = cur; cur = alt; alt = tmp;
    }

    k_nhwc2nchw<<<nblk64, 256, 0, stream>>>(cur, (float*)d_out);
}
